// BidirectionalLSTM_55508157334110
// MI455X (gfx1250) — compile-verified, matrix operand order checked by reading, not run on hardware
//
#include <hip/hip_runtime.h>
#include <hip/hip_bf16.h>

// ---------------------------------------------------------------------------
// Bidirectional LSTM, CDNA5 (gfx1250) design:
//   * one-hot input => column gather, precomputed per step (bf16)
//   * recurrent weights permuted to row' = 4*j + gate, converted to bf16 once;
//     both matrices (64 MB) stay resident in the 192 MB L2
//   * persistent kernel, 64 WGs x 512 threads, software grid barrier per step
//   * mat-vec via v_wmma_f32_16x16x32_bf16 with h broadcast across B columns
//   * final projection = bf16 WMMA GEMM [2048x4096]x[4096x1024]
// ---------------------------------------------------------------------------

#define TT 2048
#define VV 1024
#define HH 2048
#define FH 8192            // 4*H
#define IN_W (VV + HH)     // 3072, row stride of Wf_w / Wb_w

typedef __attribute__((ext_vector_type(16))) __bf16 v16bf;
typedef __attribute__((ext_vector_type(8)))  __bf16 v8bf;
typedef __attribute__((ext_vector_type(8)))  float  v8f;

__device__ __forceinline__ unsigned short f2bf(float f) {
    unsigned int u = __float_as_uint(f);
    unsigned int r = u + 0x7FFFu + ((u >> 16) & 1u);   // round-to-nearest-even
    return (unsigned short)(r >> 16);
}
__device__ __forceinline__ float bf2f(unsigned short h) {
    return __uint_as_float(((unsigned int)h) << 16);
}

// pack two 16-byte bf16 chunks into one v16bf WMMA operand
__device__ __forceinline__ v16bf pack16(v8bf lo, v8bf hi) {
    v16bf r;
#pragma unroll
    for (int q = 0; q < 8; ++q) { r[q] = lo[q]; r[q + 8] = hi[q]; }
    return r;
}

// ---------------------------------------------------------------------------
// software grid barrier (persistent kernel; all 64 WGs co-resident)
// ---------------------------------------------------------------------------
__device__ __forceinline__ void grid_barrier(int* cnt, volatile int* gen, int nwg) {
    __syncthreads();
    if (threadIdx.x == 0) {
        __threadfence();
        int g = *gen;
        if (atomicAdd(cnt, 1) == nwg - 1) {
            *cnt = 0;
            __threadfence();
            atomicAdd((int*)gen, 1);
        } else {
            while (*gen == g) { __builtin_amdgcn_s_sleep(2); }
        }
        __threadfence();
    }
    __syncthreads();
}

// ---------------------------------------------------------------------------
// init: zero h buffers, c state, barrier
// ---------------------------------------------------------------------------
__global__ void k_init(unsigned short* hcur, float* cst, int* bar) {
    int i = blockIdx.x * blockDim.x + threadIdx.x;
    if (i < 2 * 2 * HH) hcur[i] = 0;
    if (i < 2 * HH)     cst[i]  = 0.f;
    if (i < 2)          bar[i]  = 0;
}

// ---------------------------------------------------------------------------
// recurrent weight slice -> bf16, rows permuted to 4*j + gate
// ---------------------------------------------------------------------------
__global__ void k_prep_wh(const float* __restrict__ Wf, const float* __restrict__ Wb,
                          unsigned short* __restrict__ Whp) {
    long i = (long)blockIdx.x * blockDim.x + threadIdx.x;   // over 2*FH*HH
    if (i >= 2L * FH * HH) return;
    int  k   = (int)(i % HH);
    long r   = i / HH;
    int  rp  = (int)(r % FH);
    int  dir = (int)(r / FH);
    int  g = rp & 3, j = rp >> 2;
    const float* W = dir ? Wb : Wf;
    Whp[i] = f2bf(W[(size_t)(g * HH + j) * IN_W + VV + k]);
}

// ---------------------------------------------------------------------------
// one-hot input: inp[dir][t][row'] = W[orig_row][idx[t]] + b[orig_row]  (bf16)
// ---------------------------------------------------------------------------
__global__ void k_gather(const float* __restrict__ Wf, const float* __restrict__ bf,
                         const float* __restrict__ Wb, const float* __restrict__ bb,
                         const int* __restrict__ idx, unsigned short* __restrict__ inp) {
    long i = (long)blockIdx.x * blockDim.x + threadIdx.x;   // over 2*TT*FH
    if (i >= 2L * TT * FH) return;
    int  rp  = (int)(i % FH);
    long t2  = i / FH;
    int  t   = (int)(t2 % TT);
    int  dir = (int)(t2 / TT);
    int  g = rp & 3, j = rp >> 2;
    int  orig = g * HH + j;
    const float* W = dir ? Wb : Wf;
    const float* b = dir ? bb : bf;
    int c = idx[t];
    inp[i] = f2bf(W[(size_t)orig * IN_W + c] + b[orig]);
}

// ---------------------------------------------------------------------------
// Wo -> bf16
// ---------------------------------------------------------------------------
__global__ void k_prep_wo(const float* __restrict__ Wo, unsigned short* __restrict__ Wob) {
    long i = (long)blockIdx.x * blockDim.x + threadIdx.x;   // over VV*2*HH
    if (i >= (long)VV * 2 * HH) return;
    Wob[i] = f2bf(Wo[i]);
}

// ---------------------------------------------------------------------------
// persistent bidirectional LSTM recurrence
//   WGs 0..31: forward (t = 0..T-1), WGs 32..63: backward (t = T-1..0)
//   each WG: 256 permuted gate rows = 64 h-indices; 16 waves x 16-row tiles
// ---------------------------------------------------------------------------
__global__ void __launch_bounds__(512, 1)
k_lstm(const unsigned short* __restrict__ Whp, const unsigned short* __restrict__ inp,
       unsigned short* hcur, unsigned short* __restrict__ hs, float* cst, int* bar) {
    const int NWG = 64;
    int wg   = blockIdx.x;
    int dir  = wg >> 5;          // 0 fwd, 1 bwd
    int wgd  = wg & 31;
    int lane = threadIdx.x & 31;
    int wave = threadIdx.x >> 5; // 0..15

    const unsigned short* Wh = Whp + (size_t)dir * FH * HH;

    __shared__ __align__(16) unsigned short hsm[HH]; // h_{t-1} staged in LDS
    __shared__ float gsm[256];                       // this WG's gate pre-acts

    int row0 = wgd * 256 + wave * 16;                // permuted gate-row tile
    int koff = (lane & 16) ? 8 : 0;                  // A/B K-half per lane group
    const unsigned short* arow = Wh + (size_t)(row0 + (lane & 15)) * HH;

    for (int t = 0; t < TT; ++t) {
        // stage previous h (bf16, 4 KB) into LDS
        const unsigned short* hrd = hcur + ((size_t)(t & 1) * 2 + dir) * HH;
        for (int i = threadIdx.x; i < HH; i += 512) hsm[i] = hrd[i];
        __syncthreads();

        // gates_tile = W_h_tile @ h   (B = h broadcast into all 16 columns)
        v8f acc = {};
        for (int kc = 0; kc < HH; kc += 32) {
            v8bf al = *(const v8bf*)(arow + kc + koff);
            v8bf ah = *(const v8bf*)(arow + kc + 16 + koff);
            v8bf bl = *(const v8bf*)(&hsm[kc + koff]);
            v8bf bh = *(const v8bf*)(&hsm[kc + 16 + koff]);
            acc = __builtin_amdgcn_wmma_f32_16x16x32_bf16(
                false, pack16(al, ah), false, pack16(bl, bh),
                (short)0, acc, false, false);
        }
        // columns are identical; lane 0 holds rows 0..7, lane 16 rows 8..15
        if ((lane & 15) == 0) {
            int rb = wave * 16 + (lane >> 4) * 8;
#pragma unroll
            for (int q = 0; q < 8; ++q) gsm[rb + q] = acc[q];
        }
        __syncthreads();

        // element-wise cell update for this WG's 64 h-indices
        int tt = dir ? (TT - 1 - t) : t;
        if (threadIdx.x < 64) {
            int j    = threadIdx.x;
            int hidx = wgd * 64 + j;
            const unsigned short* ip =
                inp + ((size_t)dir * TT + tt) * FH + wgd * 256 + 4 * j;
            float gi = gsm[4 * j + 0] + bf2f(ip[0]);
            float gf = gsm[4 * j + 1] + bf2f(ip[1]);
            float gg = gsm[4 * j + 2] + bf2f(ip[2]);
            float go = gsm[4 * j + 3] + bf2f(ip[3]);
            gi = 1.f / (1.f + __expf(-gi));
            gf = 1.f / (1.f + __expf(-gf));
            go = 1.f / (1.f + __expf(-go));
            gg = tanhf(gg);
            float c  = cst[dir * HH + hidx];
            float cn = gf * c + gi * gg;
            cst[dir * HH + hidx] = cn;
            float hn = go * tanhf(cn);
            unsigned short hb = f2bf(hn);
            hcur[((size_t)((t + 1) & 1) * 2 + dir) * HH + hidx] = hb; // next-step buf
            hs[((size_t)dir * TT + tt) * HH + hidx] = hb;             // history
        }
        grid_barrier(bar, bar + 1, NWG);
    }
}

// ---------------------------------------------------------------------------
// out[T][V] = [hf|hb] @ Wo^T + Wo_b   — bf16 WMMA GEMM, one wave per 16x16 tile
// ---------------------------------------------------------------------------
__global__ void __launch_bounds__(256)
k_out(const unsigned short* __restrict__ hs, const unsigned short* __restrict__ Wob,
      const float* __restrict__ Wo_b, float* __restrict__ out) {
    int wave = threadIdx.x >> 5, lane = threadIdx.x & 31;
    int tile = blockIdx.x * 8 + wave;      // 8192 tiles = 128 x 64
    int tm = tile >> 6, tn = tile & 63;
    int m0 = tm * 16, n0 = tn * 16;
    int koff = (lane & 16) ? 8 : 0;

    const unsigned short* brow = Wob + (size_t)(n0 + (lane & 15)) * (2 * HH);

    v8f acc = {};
    for (int kc = 0; kc < 2 * HH; kc += 32) {
        const unsigned short* hb = hs + (kc < HH ? 0 : (size_t)TT * HH);
        int k = (kc < HH) ? kc : (kc - HH);
        const unsigned short* aptr = hb + (size_t)(m0 + (lane & 15)) * HH + k;
        v8bf al = *(const v8bf*)(aptr + koff);
        v8bf ah = *(const v8bf*)(aptr + 16 + koff);
        v8bf bl = *(const v8bf*)(brow + kc + koff);
        v8bf bh = *(const v8bf*)(brow + kc + 16 + koff);
        acc = __builtin_amdgcn_wmma_f32_16x16x32_bf16(
            false, pack16(al, ah), false, pack16(bl, bh),
            (short)0, acc, false, false);
    }
    int n  = n0 + (lane & 15);
    int mb = m0 + (lane >> 4) * 8;
    float bias = Wo_b[n];
#pragma unroll
    for (int q = 0; q < 8; ++q)
        out[(size_t)(mb + q) * VV + n] = acc[q] + bias;
}

// ---------------------------------------------------------------------------
extern "C" void kernel_launch(void* const* d_in, const int* in_sizes, int n_in,
                              void* d_out, int out_size, void* d_ws, size_t ws_size,
                              hipStream_t stream) {
    const float* Wf_w = (const float*)d_in[0];
    const float* Wf_b = (const float*)d_in[1];
    const float* Wb_w = (const float*)d_in[2];
    const float* Wb_b = (const float*)d_in[3];
    const float* Wo_w = (const float*)d_in[4];
    const float* Wo_b = (const float*)d_in[5];
    const int*   idx  = (const int*)d_in[6];
    float* out = (float*)d_out;

    char* ws = (char*)d_ws;
    size_t o = 0;
    unsigned short* Whp  = (unsigned short*)(ws + o); o += (size_t)2 * FH * HH * 2;  // 64 MB
    unsigned short* inp  = (unsigned short*)(ws + o); o += (size_t)2 * TT * FH * 2;  // 64 MB
    unsigned short* hsb  = (unsigned short*)(ws + o); o += (size_t)2 * TT * HH * 2;  // 16 MB
    unsigned short* Wob  = (unsigned short*)(ws + o); o += (size_t)VV * 2 * HH * 2;  //  8 MB
    unsigned short* hcur = (unsigned short*)(ws + o); o += (size_t)2 * 2 * HH * 2;
    float*          cst  = (float*)(ws + o);          o += (size_t)2 * HH * 4;
    int*            bar  = (int*)(ws + o);            o += 256;

    k_init   <<<32,     256, 0, stream>>>(hcur, cst, bar);
    k_prep_wh<<<131072, 256, 0, stream>>>(Wf_w, Wb_w, Whp);
    k_gather <<<131072, 256, 0, stream>>>(Wf_w, Wf_b, Wb_w, Wb_b, idx, inp);
    k_prep_wo<<<16384,  256, 0, stream>>>(Wo_w, Wob);
    k_lstm   <<<64,     512, 0, stream>>>(Whp, inp, hcur, hsb, cst, bar);
    k_out    <<<1024,   256, 0, stream>>>(hsb, Wob, Wo_b, out);
}
